// TransformerBlock_49331994362545
// MI455X (gfx1250) — matrix operand order repaired and checked by reading, not compiled, not run on hardware
//
#include <hip/hip_runtime.h>

// ---------------------------------------------------------------------------
// MI455X (gfx1250, wave32) implementation.
// All matmuls go through one bf16 WMMA GEMM kernel (v_wmma_f32_16x16x32_bf16),
// weights converted+transposed to bf16 once per launch (HBM-bound workload:
// halving weight bytes matters more than anything else on a 23.3 TB/s part).
// Round 2: 2x2 register blocking per wave (32x32 tile, 4 accumulators) ->
// ~21 flop/byte from L2 and 4 independent WMMA chains per load group.
// ---------------------------------------------------------------------------

typedef __attribute__((ext_vector_type(16))) __bf16 v16bf;
typedef __attribute__((ext_vector_type(8)))  float  v8f;

#define Bc   2
#define Tc   1024
#define Dc   1024
#define HQc  16
#define HKVc 4
#define KDc  64
#define VDc  64
#define Gc   4
#define Ec   8
#define CAPc 1024
#define HIDc 2048
#define NTOK (Bc * Tc)

union FragBF {
    v16bf v;
    uint4 q[2];
};

// ---------------------------------------------------------------------------
// Generic batched GEMM: C[M,N] (f32) = A[M,K] (bf16, row-major) * BT[N,K]^T
// (bf16, row-major i.e. B pre-transposed), optional residual R added (Z==1).
// Block = 4 waves; block tile 64x64; each wave computes a 32x32 tile with
// 2x2 WMMA register blocking. M, N must be multiples of 64 (true for all
// call sites: M in {2048,1024}, N in {1024,256,64,2048} with 64x64 grid).
// ---------------------------------------------------------------------------
__global__ __launch_bounds__(128)
void gemm_bf16_kernel(const __bf16* __restrict__ A,  long long sAz, int adiv, int lda,
                      const __bf16* __restrict__ BT, long long sBz, int bdiv, int ldb,
                      float* __restrict__ C, long long sCz, long long sCz2,
                      int cdiv, int cmod, int ldc,
                      const float* __restrict__ R, int ldr,
                      int M, int N, int K)
{
    const int z = blockIdx.z;
    A  += (long long)(z / adiv) * sAz;
    BT += (long long)(z / bdiv) * sBz;
    const long long coff = (long long)(z / cdiv) * sCz + (long long)(z % cmod) * sCz2;

    const int lane = threadIdx.x;                 // 0..31
    const int wy = threadIdx.y & 1, wx = threadIdx.y >> 1;
    const int mbase = blockIdx.x * 64 + wy * 32;
    const int nbase = blockIdx.y * 64 + wx * 32;
    if (mbase >= M || nbase >= N) return;

    const int idx  = lane & 15;                   // M row (A) / N row (BT)
    const int half = lane >> 4;                   // K-half selector

    // A frag: lane half 0 -> K = k+0..7 & k+16..23 ; half 1 -> +8 / +24
    const __bf16* ap0 = A + (long long)(mbase + idx) * lda + half * 8;
    const __bf16* ap1 = ap0 + (long long)16 * lda;
    // B frag: lane half 0 -> K = k+0..7 & k+16..23 ; half 1 -> +8 / +24
    const __bf16* bp0 = BT + (long long)(nbase + idx) * ldb + half * 8;
    const __bf16* bp1 = bp0 + (long long)16 * ldb;

    v8f c00 = {}, c01 = {}, c10 = {}, c11 = {};
    for (int k = 0; k < K; k += 32) {
        FragBF a0, a1, b0, b1;
        a0.q[0] = *(const uint4*)(ap0 + k);
        a0.q[1] = *(const uint4*)(ap0 + k + 16);
        a1.q[0] = *(const uint4*)(ap1 + k);
        a1.q[1] = *(const uint4*)(ap1 + k + 16);
        b0.q[0] = *(const uint4*)(bp0 + k);
        b0.q[1] = *(const uint4*)(bp0 + k + 16);
        b1.q[0] = *(const uint4*)(bp1 + k);
        b1.q[1] = *(const uint4*)(bp1 + k + 16);
        __builtin_prefetch(ap0 + k + 256, 0, 1);   // global_prefetch_b8
        __builtin_prefetch(bp0 + k + 256, 0, 1);
        c00 = __builtin_amdgcn_wmma_f32_16x16x32_bf16(
                  false, a0.v, false, b0.v, (short)0, c00, false, false);
        c01 = __builtin_amdgcn_wmma_f32_16x16x32_bf16(
                  false, a0.v, false, b1.v, (short)0, c01, false, false);
        c10 = __builtin_amdgcn_wmma_f32_16x16x32_bf16(
                  false, a1.v, false, b0.v, (short)0, c10, false, false);
        c11 = __builtin_amdgcn_wmma_f32_16x16x32_bf16(
                  false, a1.v, false, b1.v, (short)0, c11, false, false);
    }

    // f32 C/D layout: lane half 0 -> rows 0..7, half 1 -> rows 8..15; col = idx
    auto store_tile = [&](const v8f& acc, int mb, int nb) {
        float* cp = C + coff + (long long)(mb + half * 8) * ldc + nb + idx;
        if (R) {
            const float* rp = R + (long long)(mb + half * 8) * ldr + nb + idx;
            for (int r = 0; r < 8; ++r)
                cp[(long long)r * ldc] = acc[r] + rp[(long long)r * ldr];
        } else {
            for (int r = 0; r < 8; ++r)
                cp[(long long)r * ldc] = acc[r];
        }
    };
    store_tile(c00, mbase,      nbase);
    store_tile(c01, mbase,      nbase + 16);
    store_tile(c10, mbase + 16, nbase);
    store_tile(c11, mbase + 16, nbase + 16);
}

// ---------------------------------------------------------------------------
// Weight transpose + fp32 -> bf16 convert:  W[K,N] -> WT[N,K]  (batched over z)
// ---------------------------------------------------------------------------
__global__ __launch_bounds__(256)
void wtrans_kernel(const float* __restrict__ W, __bf16* __restrict__ WT,
                   int K, int N, long long inS, long long outS)
{
    const long long z = blockIdx.y;
    const long long i = (long long)blockIdx.x * 256 + threadIdx.x;
    if (i < (long long)K * N) {
        const int k = (int)(i / N), n = (int)(i % N);
        WT[z * outS + (long long)n * K + k] = (__bf16)W[z * inS + i];
    }
}

__global__ __launch_bounds__(256)
void f2b_kernel(const float* __restrict__ X, __bf16* __restrict__ Y, long long n)
{
    const long long i = (long long)blockIdx.x * 256 + threadIdx.x;
    if (i < n) Y[i] = (__bf16)X[i];
}

// ---------------------------------------------------------------------------
// RMSNorm: one block per row; writes f32 (optional) and bf16 copies.
// ---------------------------------------------------------------------------
__global__ __launch_bounds__(256)
void rmsnorm_kernel(const float* __restrict__ X, const float* __restrict__ scale,
                    float* __restrict__ Yf, __bf16* __restrict__ Yb)
{
    const long long row = blockIdx.x;
    const float* x = X + row * Dc;
    __shared__ float red[256];
    float ss = 0.f;
    for (int d = threadIdx.x; d < Dc; d += 256) { float v = x[d]; ss += v * v; }
    red[threadIdx.x] = ss; __syncthreads();
    for (int o = 128; o > 0; o >>= 1) {
        if (threadIdx.x < o) red[threadIdx.x] += red[threadIdx.x + o];
        __syncthreads();
    }
    const float r = rsqrtf(red[0] / (float)Dc + 1e-6f);
    for (int d = threadIdx.x; d < Dc; d += 256) {
        const float v = x[d] * r * scale[d];
        if (Yf) Yf[row * Dc + d] = v;
        Yb[row * Dc + d] = (__bf16)v;
    }
}

// ---------------------------------------------------------------------------
// RoPE on q,k (tiled-ticks convention) + repack to (b,h,t,kd) bf16;
// v converted to transposed (b,h,vd,t) bf16 so P·V can use the same GEMM.
// ---------------------------------------------------------------------------
__global__ __launch_bounds__(256)
void rope_pack_kernel(const float* __restrict__ qf, const float* __restrict__ kf,
                      const float* __restrict__ vf,
                      __bf16* __restrict__ qb, __bf16* __restrict__ kb,
                      __bf16* __restrict__ vt)
{
    const int t = blockIdx.x, b = blockIdx.y;
    const long long rq = ((long long)b * Tc + t) * (HQc * KDc);
    for (int i = threadIdx.x; i < HQc * KDc; i += 256) {
        const int h = i >> 6, kd = i & 63, j = kd & 31;
        const float theta = __powf(10000.f, -(float)(2 * j) / 64.f);
        float s, c; __sincosf((float)t * theta, &s, &c);
        const float x0 = qf[rq + i];
        const float xo = (kd < 32) ? -qf[rq + i + 32] : qf[rq + i - 32];
        qb[(((long long)(b * HQc + h)) * Tc + t) * KDc + kd] = (__bf16)(x0 * c + xo * s);
    }
    const long long rk = ((long long)b * Tc + t) * (HKVc * KDc);
    for (int i = threadIdx.x; i < HKVc * KDc; i += 256) {
        const int h = i >> 6, kd = i & 63, j = kd & 31;
        const float theta = __powf(10000.f, -(float)(2 * j) / 64.f);
        float s, c; __sincosf((float)t * theta, &s, &c);
        const float x0 = kf[rk + i];
        const float xo = (kd < 32) ? -kf[rk + i + 32] : kf[rk + i - 32];
        kb[(((long long)(b * HKVc + h)) * Tc + t) * KDc + kd] = (__bf16)(x0 * c + xo * s);
    }
    const long long rv = ((long long)b * Tc + t) * (HKVc * VDc);
    for (int i = threadIdx.x; i < HKVc * VDc; i += 256) {
        const int h = i >> 6, vd = i & 63;
        vt[(((long long)(b * HKVc + h)) * VDc + vd) * Tc + t] = (__bf16)vf[rv + i];
    }
}

// ---------------------------------------------------------------------------
// Causal softmax over a score row (scale 1/sqrt(64)=0.125); writes bf16 probs.
// ---------------------------------------------------------------------------
__global__ __launch_bounds__(256)
void softmax_causal_kernel(const float* __restrict__ S, __bf16* __restrict__ P)
{
    const int t = blockIdx.x;
    const long long z = blockIdx.y;
    const float* row = S + (z * Tc + t) * (long long)Tc;
    __bf16* prow = P + (z * Tc + t) * (long long)Tc;
    __shared__ float red[256];
    float mx = -1e30f;
    for (int s = threadIdx.x; s < Tc; s += 256)
        if (s <= t) mx = fmaxf(mx, row[s] * 0.125f);
    red[threadIdx.x] = mx; __syncthreads();
    for (int o = 128; o > 0; o >>= 1) {
        if (threadIdx.x < o) red[threadIdx.x] = fmaxf(red[threadIdx.x], red[threadIdx.x + o]);
        __syncthreads();
    }
    mx = red[0]; __syncthreads();
    float sum = 0.f;
    for (int s = threadIdx.x; s < Tc; s += 256)
        if (s <= t) sum += __expf(row[s] * 0.125f - mx);
    red[threadIdx.x] = sum; __syncthreads();
    for (int o = 128; o > 0; o >>= 1) {
        if (threadIdx.x < o) red[threadIdx.x] += red[threadIdx.x + o];
        __syncthreads();
    }
    const float inv = 1.f / red[0];
    for (int s = threadIdx.x; s < Tc; s += 256) {
        const float v = (s <= t) ? __expf(row[s] * 0.125f - mx) * inv : 0.f;
        prow[s] = (__bf16)v;
    }
}

// ---------------------------------------------------------------------------
// Router: fp32 logits, top-2 softmax gates, sequential capacity assignment.
// Single block (routing is <0.1% of FLOPs; determinism matters more).
// ---------------------------------------------------------------------------
__global__ __launch_bounds__(256)
void router_kernel(const float* __restrict__ h2f, const float* __restrict__ rw,
                   const float* __restrict__ rb,
                   float* __restrict__ logits, int* __restrict__ assign,
                   float* __restrict__ gate, int* __restrict__ expidx,
                   int* __restrict__ posidx, int* __restrict__ tmask,
                   int* __restrict__ slot_token)
{
    for (int idx = threadIdx.x; idx < NTOK * Ec; idx += 256) {
        const int tok = idx >> 3, e = idx & 7;
        const float* x = h2f + (long long)tok * Dc;
        float acc = rb[e];
        for (int d = 0; d < Dc; ++d) acc += x[d] * rw[d * Ec + e];
        logits[idx] = acc;
    }
    for (int i = threadIdx.x; i < Ec * CAPc; i += 256) slot_token[i] = -1;
    __syncthreads();
    for (int tok = threadIdx.x; tok < NTOK; tok += 256) {
        const float* lg = logits + tok * Ec;
        int b0 = 0; float v0 = lg[0];
        for (int e = 1; e < Ec; ++e) if (lg[e] > v0) { v0 = lg[e]; b0 = e; }
        int b1 = -1; float v1 = -3.4e38f;
        for (int e = 0; e < Ec; ++e) if (e != b0 && lg[e] > v1) { v1 = lg[e]; b1 = e; }
        const float m = fmaxf(v0, v1);
        const float e0 = __expf(v0 - m), e1 = __expf(v1 - m);
        const float inv = 1.f / (e0 + e1);
        assign[tok * 2 + 0] = b0; assign[tok * 2 + 1] = b1;
        gate[tok * 2 + 0] = e0 * inv; gate[tok * 2 + 1] = e1 * inv;
    }
    __syncthreads();
    if (threadIdx.x == 0) {
        int counts[Ec];
        for (int e = 0; e < Ec; ++e) counts[e] = 0;
        for (int tok = 0; tok < NTOK; ++tok)
            for (int k = 0; k < 2; ++k) {
                const int e = assign[tok * 2 + k];
                const int p = counts[e]++;
                if (p < CAPc) {
                    slot_token[e * CAPc + p] = tok;
                    expidx[tok * 2 + k] = e; posidx[tok * 2 + k] = p;
                    tmask[tok * 2 + k] = 1;
                } else {
                    expidx[tok * 2 + k] = 0; posidx[tok * 2 + k] = 0;
                    tmask[tok * 2 + k] = 0;
                }
            }
    }
}

__global__ __launch_bounds__(256)
void dispatch_kernel(const int* __restrict__ slot_token,
                     const __bf16* __restrict__ h2b, __bf16* __restrict__ grouped)
{
    const int row = blockIdx.x;                 // e*CAP + p
    const int tok = slot_token[row];
    __bf16* g = grouped + (long long)row * Dc;
    if (tok >= 0) {
        const __bf16* src = h2b + (long long)tok * Dc;
        for (int d = threadIdx.x; d < Dc; d += 256) g[d] = src[d];
    } else {
        for (int d = threadIdx.x; d < Dc; d += 256) g[d] = (__bf16)0.f;
    }
}

__global__ __launch_bounds__(256)
void gelumul_kernel(const float* __restrict__ t1, const float* __restrict__ t2,
                    __bf16* __restrict__ hh, long long n)
{
    const long long i = (long long)blockIdx.x * 256 + threadIdx.x;
    if (i < n) {
        const float x = t2[i] * t1[i];
        const float g = 0.5f * x * (1.f + tanhf(0.7978845608f * (x + 0.044715f * x * x * x)));
        hh[i] = (__bf16)g;
    }
}

__global__ __launch_bounds__(256)
void combine_kernel(const float* __restrict__ xattn, const float* __restrict__ h2f,
                    const float* __restrict__ eout, const float* __restrict__ gate,
                    const int* __restrict__ expidx, const int* __restrict__ posidx,
                    const int* __restrict__ tmask, float* __restrict__ out)
{
    const int tok = blockIdx.x;
    const int m0 = tmask[tok * 2], m1 = tmask[tok * 2 + 1];
    const float g0 = gate[tok * 2], g1 = gate[tok * 2 + 1];
    const float* r0 = m0 ? eout + ((long long)expidx[tok * 2] * CAPc + posidx[tok * 2]) * Dc
                         : h2f + (long long)tok * Dc;
    const float* r1 = m1 ? eout + ((long long)expidx[tok * 2 + 1] * CAPc + posidx[tok * 2 + 1]) * Dc
                         : h2f + (long long)tok * Dc;
    for (int d = threadIdx.x; d < Dc; d += 256)
        out[(long long)tok * Dc + d] = xattn[(long long)tok * Dc + d] + g0 * r0[d] + g1 * r1[d];
}

// ---------------------------------------------------------------------------
extern "C" void kernel_launch(void* const* d_in, const int* in_sizes, int n_in,
                              void* d_out, int out_size, void* d_ws, size_t ws_size,
                              hipStream_t stream)
{
    const float* x   = (const float*)d_in[0];
    const float* wq  = (const float*)d_in[1];
    const float* wk  = (const float*)d_in[2];
    const float* wv  = (const float*)d_in[3];
    const float* wo  = (const float*)d_in[4];
    const float* rw  = (const float*)d_in[5];
    const float* rb  = (const float*)d_in[6];
    const float* w1  = (const float*)d_in[7];
    const float* w2  = (const float*)d_in[8];
    const float* w3  = (const float*)d_in[9];
    const float* n1s = (const float*)d_in[10];
    const float* n2s = (const float*)d_in[11];
    float* out = (float*)d_out;

    char* p = (char*)d_ws;
    auto alloc = [&](size_t bytes) -> void* {
        void* r = (void*)p;
        p += (bytes + 255) & ~(size_t)255;
        return r;
    };

    // ---- persistent bf16 transposed weights (~101 MB) ----
    __bf16* wqT = (__bf16*)alloc((size_t)Dc * Dc * 2);
    __bf16* wkT = (__bf16*)alloc((size_t)Dc * (HKVc * KDc) * 2);
    __bf16* wvT = (__bf16*)alloc((size_t)Dc * (HKVc * VDc) * 2);
    __bf16* woT = (__bf16*)alloc((size_t)Dc * Dc * 2);
    __bf16* w1T = (__bf16*)alloc((size_t)Ec * Dc * HIDc * 2);
    __bf16* w2T = (__bf16*)alloc((size_t)Ec * Dc * HIDc * 2);
    __bf16* w3T = (__bf16*)alloc((size_t)Ec * HIDc * Dc * 2);
    // ---- activations ----
    __bf16* h1b   = (__bf16*)alloc((size_t)NTOK * Dc * 2);
    float*  qf    = (float*)alloc((size_t)NTOK * Dc * 4);
    float*  kf    = (float*)alloc((size_t)NTOK * HKVc * KDc * 4);
    float*  vf    = (float*)alloc((size_t)NTOK * HKVc * VDc * 4);
    __bf16* qb    = (__bf16*)alloc((size_t)NTOK * Dc * 2);
    __bf16* kb    = (__bf16*)alloc((size_t)NTOK * HKVc * KDc * 2);
    __bf16* vt    = (__bf16*)alloc((size_t)NTOK * HKVc * VDc * 2);
    float*  scores = (float*)alloc((size_t)Bc * HQc * Tc * Tc * 4);   // 128 MB (reused: t1/t2)
    __bf16* probs  = (__bf16*)alloc((size_t)Bc * HQc * Tc * Tc * 2);  //  64 MB (reused: eout/hh)
    float*  attnf  = (float*)alloc((size_t)NTOK * Dc * 4);
    __bf16* attnb  = (__bf16*)alloc((size_t)NTOK * Dc * 2);
    float*  xattn  = (float*)alloc((size_t)NTOK * Dc * 4);
    float*  h2f    = (float*)alloc((size_t)NTOK * Dc * 4);
    __bf16* h2b    = (__bf16*)alloc((size_t)NTOK * Dc * 2);
    float*  logits = (float*)alloc((size_t)NTOK * Ec * 4);
    int*    assign = (int*)alloc((size_t)NTOK * 2 * 4);
    float*  gate   = (float*)alloc((size_t)NTOK * 2 * 4);
    int*    expidx = (int*)alloc((size_t)NTOK * 2 * 4);
    int*    posidx = (int*)alloc((size_t)NTOK * 2 * 4);
    int*    tmask  = (int*)alloc((size_t)NTOK * 2 * 4);
    int*    slot_token = (int*)alloc((size_t)Ec * CAPc * 4);
    __bf16* grouped = (__bf16*)alloc((size_t)Ec * CAPc * Dc * 2);
    // phase-B aliases of the big attention buffers
    float*  t1   = scores;                                     // 8*1024*2048 f32
    float*  t2   = scores + (size_t)Ec * CAPc * HIDc;          // second half
    float*  eout = (float*)probs;                              // 8*1024*1024 f32
    __bf16* hhb  = (__bf16*)(probs + (size_t)Ec * CAPc * HIDc);// 8*1024*2048 bf16

    const dim3 gb(32, 4, 1);
    auto gemm = [&](const __bf16* A, long long sAz, int adiv, int lda,
                    const __bf16* BT, long long sBz, int bdiv, int ldb,
                    float* C, long long sCz, long long sCz2, int cdiv, int cmod, int ldc,
                    const float* R, int ldr, int M, int N, int K, int Z) {
        dim3 gg(M / 64, (N + 63) / 64, Z);
        gemm_bf16_kernel<<<gg, gb, 0, stream>>>(A, sAz, adiv, lda, BT, sBz, bdiv, ldb,
                                                C, sCz, sCz2, cdiv, cmod, ldc,
                                                R, ldr, M, N, K);
    };
    auto wtrans = [&](const float* W, __bf16* WT, int K, int N, int Z,
                      long long inS, long long outS) {
        dim3 gg((unsigned)(((long long)K * N + 255) / 256), Z);
        wtrans_kernel<<<gg, 256, 0, stream>>>(W, WT, K, N, inS, outS);
    };

    // ---- weight conversion (once per launch; HBM-bound workload => bf16) ----
    wtrans(wq, wqT, Dc, Dc, 1, 0, 0);
    wtrans(wk, wkT, Dc, HKVc * KDc, 1, 0, 0);
    wtrans(wv, wvT, Dc, HKVc * VDc, 1, 0, 0);
    wtrans(wo, woT, Dc, Dc, 1, 0, 0);
    wtrans(w1, w1T, Dc, HIDc, Ec, (long long)Dc * HIDc, (long long)Dc * HIDc);
    wtrans(w2, w2T, Dc, HIDc, Ec, (long long)Dc * HIDc, (long long)Dc * HIDc);
    wtrans(w3, w3T, HIDc, Dc, Ec, (long long)HIDc * Dc, (long long)HIDc * Dc);

    // ---- attention ----
    rmsnorm_kernel<<<NTOK, 256, 0, stream>>>(x, n1s, nullptr, h1b);
    gemm(h1b, 0, 1, Dc, wqT, 0, 1, Dc, qf, 0, 0, 1, 1, Dc, nullptr, 0,
         NTOK, Dc, Dc, 1);
    gemm(h1b, 0, 1, Dc, wkT, 0, 1, Dc, kf, 0, 0, 1, 1, HKVc * KDc, nullptr, 0,
         NTOK, HKVc * KDc, Dc, 1);
    gemm(h1b, 0, 1, Dc, wvT, 0, 1, Dc, vf, 0, 0, 1, 1, HKVc * VDc, nullptr, 0,
         NTOK, HKVc * VDc, Dc, 1);
    rope_pack_kernel<<<dim3(Tc, Bc), 256, 0, stream>>>(qf, kf, vf, qb, kb, vt);
    // S = Q K^T : z over b*HQ+h ; kv batch = z/4
    gemm(qb, (long long)Tc * KDc, 1, KDc,
         kb, (long long)Tc * KDc, Gc, KDc,
         scores, (long long)Tc * Tc, 0, 1, 1, Tc, nullptr, 0,
         Tc, Tc, KDc, Bc * HQc);
    softmax_causal_kernel<<<dim3(Tc, Bc * HQc), 256, 0, stream>>>(scores, probs);
    // attn = P V : C written at (b)*T*D + h*64 column offset
    gemm(probs, (long long)Tc * Tc, 1, Tc,
         vt, (long long)VDc * Tc, Gc, Tc,
         attnf, (long long)Tc * Dc, 64, HQc, HQc, Dc, nullptr, 0,
         Tc, VDc, Tc, Bc * HQc);
    f2b_kernel<<<(unsigned)(((size_t)NTOK * Dc + 255) / 256), 256, 0, stream>>>(
        attnf, attnb, (long long)NTOK * Dc);
    // x = x + attn @ wo   (residual fused)
    gemm(attnb, 0, 1, Dc, woT, 0, 1, Dc, xattn, 0, 0, 1, 1, Dc, x, Dc,
         NTOK, Dc, Dc, 1);

    // ---- MoE ----
    rmsnorm_kernel<<<NTOK, 256, 0, stream>>>(xattn, n2s, h2f, h2b);
    router_kernel<<<1, 256, 0, stream>>>(h2f, rw, rb, logits, assign, gate,
                                         expidx, posidx, tmask, slot_token);
    dispatch_kernel<<<Ec * CAPc, 256, 0, stream>>>(slot_token, h2b, grouped);
    gemm(grouped, (long long)CAPc * Dc, 1, Dc,
         w1T, (long long)Dc * HIDc, 1, Dc,
         t1, (long long)CAPc * HIDc, 0, 1, 1, HIDc, nullptr, 0,
         CAPc, HIDc, Dc, Ec);
    gemm(grouped, (long long)CAPc * Dc, 1, Dc,
         w2T, (long long)Dc * HIDc, 1, Dc,
         t2, (long long)CAPc * HIDc, 0, 1, 1, HIDc, nullptr, 0,
         CAPc, HIDc, Dc, Ec);
    {
        const long long n = (long long)Ec * CAPc * HIDc;
        gelumul_kernel<<<(unsigned)((n + 255) / 256), 256, 0, stream>>>(t1, t2, hhb, n);
    }
    gemm(hhb, (long long)CAPc * HIDc, 1, HIDc,
         w3T, (long long)HIDc * Dc, 1, HIDc,
         eout, (long long)CAPc * Dc, 0, 1, 1, Dc, nullptr, 0,
         CAPc, Dc, HIDc, Ec);
    combine_kernel<<<NTOK, 256, 0, stream>>>(xattn, h2f, eout, gate,
                                             expidx, posidx, tmask, out);
    (void)in_sizes; (void)n_in; (void)out_size; (void)ws_size;
}
